// HardMoE_21689584845166
// MI455X (gfx1250) — matrix operand order repaired and checked by reading, not compiled, not run on hardware
//
#include <hip/hip_runtime.h>
#include <hip/hip_bf16.h>

#define B_ROWS 8192
#define D_IN   1024
#define D_OUT  1024
#define NE     16

#define TM 64     // block rows
#define TN 128    // block cols
#define KT 32     // k per WMMA step
#define LDSTR 40  // LDS row stride (elements) for n-major B tile; multiple of 8 => 16B-aligned frags

typedef __attribute__((ext_vector_type(16))) __bf16 v16bf;
typedef __attribute__((ext_vector_type(8)))  __bf16 v8bf;
typedef __attribute__((ext_vector_type(8)))  float  v8f;

// ---------------------------------------------------------------- zero counts
__global__ void moe_zero_counts(int* __restrict__ counts) {
    if (threadIdx.x < NE) counts[threadIdx.x] = 0;
}

// ---------------------------------------------------------------- gate + route
// one wave per row; lanes stride D_IN; exact fp32 gate -> robust argmax routing
__global__ __launch_bounds__(256) void moe_gate(
    const float* __restrict__ x, const float* __restrict__ Wg,
    const float* __restrict__ bg, int* __restrict__ counts,
    int* __restrict__ rowlist)
{
    const int wave = threadIdx.x >> 5;
    const int lane = threadIdx.x & 31;
    const int row  = blockIdx.x * 8 + wave;

    float acc[NE];
#pragma unroll
    for (int e = 0; e < NE; ++e) acc[e] = 0.f;

    const float* xr = x + (size_t)row * D_IN;
    for (int d = lane; d < D_IN; d += 32) {
        const float xv = xr[d];
        const float4* w4 = (const float4*)(Wg + (size_t)d * NE);
        const float4 w0 = w4[0], w1 = w4[1], w2 = w4[2], w3 = w4[3];
        acc[0]  += xv * w0.x;  acc[1]  += xv * w0.y;
        acc[2]  += xv * w0.z;  acc[3]  += xv * w0.w;
        acc[4]  += xv * w1.x;  acc[5]  += xv * w1.y;
        acc[6]  += xv * w1.z;  acc[7]  += xv * w1.w;
        acc[8]  += xv * w2.x;  acc[9]  += xv * w2.y;
        acc[10] += xv * w2.z;  acc[11] += xv * w2.w;
        acc[12] += xv * w3.x;  acc[13] += xv * w3.y;
        acc[14] += xv * w3.z;  acc[15] += xv * w3.w;
    }
#pragma unroll
    for (int off = 16; off >= 1; off >>= 1) {
#pragma unroll
        for (int e = 0; e < NE; ++e)
            acc[e] += __shfl_xor(acc[e], off, 32);
    }
    if (lane == 0) {
        float best = acc[0] + bg[0];
        int bestE = 0;
#pragma unroll
        for (int e = 1; e < NE; ++e) {
            const float v = acc[e] + bg[e];
            if (v > best) { best = v; bestE = e; }  // first-max => matches jnp.argmax
        }
        const int pos = atomicAdd(&counts[bestE], 1);
        rowlist[bestE * B_ROWS + pos] = row;
    }
}

// ---------------------------------------------------------------- grouped GEMM
// grid: (D_OUT/TN, ceil(B/TM), NE).  Block: 256 thr = 8 waves.
// wave w: mi = w&3 -> 16-row sub-tile, ng = w>>2 -> 64-col half.
// bf16x3 split (ah*bh + ah*bl + al*bh) for ~fp32 accuracy on WMMA path.
__global__ __launch_bounds__(256) void moe_expert_gemm(
    const float* __restrict__ x, const float* __restrict__ We,
    const float* __restrict__ be, const int* __restrict__ counts,
    const int* __restrict__ rowlist, float* __restrict__ out)
{
    __shared__ __attribute__((aligned(16))) __bf16 ldsH[TN * LDSTR];
    __shared__ __attribute__((aligned(16))) __bf16 ldsL[TN * LDSTR];

    const int e = blockIdx.z;
    const int cnt = counts[e];
    const int rowBase = blockIdx.y * TM;
    if (rowBase >= cnt) return;                // uniform early exit
    const int n0 = blockIdx.x * TN;

    const int tid  = threadIdx.x;
    const int wave = tid >> 5;
    const int lane = tid & 31;
    const int half = lane >> 4;
    const int l16  = lane & 15;
    const int mi   = wave & 3;
    const int ng   = wave >> 2;

    const int* rl = rowlist + e * B_ROWS;

    // gathered row for this lane's A-fragment slot (clamped for ragged tail)
    const int gRowA = rowBase + mi * 16 + l16;
    const int rowA  = rl[min(gRowA, cnt - 1)];
    const float* xA = x + (size_t)rowA * D_IN;

    v8f acc[4];
#pragma unroll
    for (int ns = 0; ns < 4; ++ns) {
#pragma unroll
        for (int i = 0; i < 8; ++i) acc[ns][i] = 0.f;
    }

    const float* Wexp = We + (size_t)e * D_IN * D_OUT;

    for (int k0 = 0; k0 < D_IN; k0 += KT) {
        __syncthreads();
        // stage We[k0..k0+32) x [n0..n0+128) -> LDS n-major, split hi/lo bf16.
        // coalesced: each wave reads one 512B k-row chunk per pass.
#pragma unroll
        for (int i = 0; i < 4; ++i) {
            const int k = i * 8 + (tid >> 5);
            const int n = (tid & 31) * 4;
            const float4 v = *(const float4*)(Wexp + (size_t)(k0 + k) * D_OUT + n0 + n);
            const float vv[4] = { v.x, v.y, v.z, v.w };
#pragma unroll
            for (int j = 0; j < 4; ++j) {
                const __bf16 h = (__bf16)vv[j];
                const __bf16 l = (__bf16)(vv[j] - (float)h);
                ldsH[(n + j) * LDSTR + k] = h;
                ldsL[(n + j) * LDSTR + k] = l;
            }
        }
        __syncthreads();

        // A fragment (16x32 bf16 layout): elems contiguous K = k0 + half*16 + i
        const float* pa = xA + k0 + half * 16;
        const float4 a0 = *(const float4*)(pa);
        const float4 a1 = *(const float4*)(pa + 4);
        const float4 a2 = *(const float4*)(pa + 8);
        const float4 a3 = *(const float4*)(pa + 12);
        const float af[16] = { a0.x, a0.y, a0.z, a0.w, a1.x, a1.y, a1.z, a1.w,
                               a2.x, a2.y, a2.z, a2.w, a3.x, a3.y, a3.z, a3.w };
        v16bf ah, al;
#pragma unroll
        for (int i = 0; i < 16; ++i) {
            const __bf16 h = (__bf16)af[i];
            ah[i] = h;
            al[i] = (__bf16)(af[i] - (float)h);
        }

#pragma unroll
        for (int ns = 0; ns < 4; ++ns) {
            const int nLoc = ng * 64 + ns * 16 + l16;
            // B fragment (32x16): lane = n, elems contiguous K = half*16 + i
            const v8bf* bhp = (const v8bf*)&ldsH[nLoc * LDSTR + half * 16];
            const v8bf* blp = (const v8bf*)&ldsL[nLoc * LDSTR + half * 16];
            const v8bf bh0 = bhp[0], bh1 = bhp[1];
            const v8bf bl0 = blp[0], bl1 = blp[1];
            v16bf bh, bl;
#pragma unroll
            for (int i = 0; i < 8; ++i) {
                bh[i] = bh0[i]; bh[i + 8] = bh1[i];
                bl[i] = bl0[i]; bl[i + 8] = bl1[i];
            }
            acc[ns] = __builtin_amdgcn_wmma_f32_16x16x32_bf16(
                false, ah, false, bh, (short)0, acc[ns], false, false);
            acc[ns] = __builtin_amdgcn_wmma_f32_16x16x32_bf16(
                false, ah, false, bl, (short)0, acc[ns], false, false);
            acc[ns] = __builtin_amdgcn_wmma_f32_16x16x32_bf16(
                false, al, false, bh, (short)0, acc[ns], false, false);
        }
    }

    // C/D layout: VGPR r -> M = r + 8*half, N = l16. Masked scatter + bias.
    int  rowOut[8];
    bool vOut[8];
#pragma unroll
    for (int r = 0; r < 8; ++r) {
        const int gM = rowBase + mi * 16 + half * 8 + r;
        vOut[r]  = (gM < cnt);
        rowOut[r] = rl[min(gM, cnt - 1)];
    }
#pragma unroll
    for (int ns = 0; ns < 4; ++ns) {
        const int n = n0 + ng * 64 + ns * 16 + l16;
        const float bias = be[e * D_OUT + n];
#pragma unroll
        for (int r = 0; r < 8; ++r) {
            if (vOut[r])
                out[(size_t)rowOut[r] * D_OUT + n] = acc[ns][r] + bias;
        }
    }
}

// ---------------------------------------------------------------- launch
extern "C" void kernel_launch(void* const* d_in, const int* in_sizes, int n_in,
                              void* d_out, int out_size, void* d_ws, size_t ws_size,
                              hipStream_t stream) {
    (void)in_sizes; (void)n_in; (void)out_size; (void)ws_size;
    const float* x  = (const float*)d_in[0];
    const float* Wg = (const float*)d_in[1];
    const float* bg = (const float*)d_in[2];
    const float* We = (const float*)d_in[3];
    const float* be = (const float*)d_in[4];
    float* out = (float*)d_out;

    int* counts  = (int*)d_ws;       // 16 ints
    int* rowlist = counts + 16;      // NE * B_ROWS ints (~512 KB)

    moe_zero_counts<<<1, 32, 0, stream>>>(counts);
    moe_gate<<<B_ROWS / 8, 256, 0, stream>>>(x, Wg, bg, counts, rowlist);

    dim3 grid(D_OUT / TN, (B_ROWS + TM - 1) / TM, NE);
    moe_expert_gemm<<<grid, 256, 0, stream>>>(x, We, be, counts, rowlist, out);
}
